// Convolution_2911987827240
// MI455X (gfx1250) — matrix operand order repaired and checked by reading, not compiled, not run on hardware
//
#include <hip/hip_runtime.h>

typedef __attribute__((ext_vector_type(16))) _Float16 v16h;
typedef __attribute__((ext_vector_type(8)))  _Float16 v8h;
typedef __attribute__((ext_vector_type(8)))  float    v8f;

namespace {
constexpr int kT      = 3072;
constexpr int kC      = 512;
constexpr int kB      = 32;
constexpr int M_BLK   = 128;
constexpr int N_CHUNK = 128;   // output-column slice per (nc) pass
constexpr int K_HALF  = 256;   // K columns per B slice (double-buffered)
constexpr int LDA     = 520;   // A LDS stride (f16): 1040B -> 4-bank rotate, conflict-free
constexpr int LDB     = 264;   // B LDS stride (f16):  528B -> 4-bank rotate, conflict-free
constexpr int N_SEG   = 8;     // (512/N_CHUNK) * (512/K_HALF) segments
}

// ---- prep: fc_w fp32 -> f16 (stays resident in the 192MB L2) ----
__global__ __launch_bounds__(256) void cvt_fcw_f16(const float* __restrict__ w,
                                                   _Float16* __restrict__ o) {
  int i = blockIdx.x * blockDim.x + threadIdx.x;
  o[i] = (_Float16)w[i];
}

// Issue one B slice (128 rows x 256 K-cols, f16) as async global->LDS DMA.
// Each wave issues exactly 16 GLOBAL_LOAD_ASYNC_TO_LDS_B128 (ASYNCcnt += 16).
__device__ __forceinline__ void stage_b_async(const _Float16* __restrict__ src,
                                              _Float16* dst, int tid) {
#pragma unroll
  for (int it = 0; it < 16; ++it) {
    const int idx = tid + it * 256;
    const int j   = idx >> 5;          // fc_w row within slice (0..127)
    const int kq  = idx & 31;          // 16B segment within 256-col slice
    const unsigned ldsoff = (unsigned)(uintptr_t)(dst + j * LDB + kq * 8);
    const unsigned voff   = (unsigned)((j * kC + kq * 8) * sizeof(_Float16));
    asm volatile("global_load_async_to_lds_b128 %0, %1, %2"
                 :: "v"(ldsoff), "v"(voff), "s"(src)
                 : "memory");
  }
}

// ---- fused: depthwise conv (fp32 sliding window) + WMMA GEMM + bias ----
__global__ __launch_bounds__(256) void fused_dwconv_gemm(
    const float* __restrict__ x,
    const float* __restrict__ w_rtg, const float* __restrict__ b_rtg,
    const float* __restrict__ w_obs, const float* __restrict__ b_obs,
    const float* __restrict__ w_act, const float* __restrict__ b_act,
    const _Float16* __restrict__ wf16,
    const float* __restrict__ fc_b,
    float* __restrict__ out)
{
  extern __shared__ _Float16 smem[];
  _Float16* lA  = smem;                        // [M_BLK][LDA]
  _Float16* lB0 = smem + M_BLK * LDA;          // [N_CHUNK][LDB] double-buffered
  _Float16* lB1 = lB0 + N_CHUNK * LDB;

  const int tid  = threadIdx.x;
  const int t0   = blockIdx.x * M_BLK;         // flat row in [0, B*T)
  const int tloc = t0 % kT;

  // Prefetch B segment 0 (nc=0, kb=0) into buf0 -- hidden behind the conv phase.
  stage_b_async(wf16, lB0, tid);

  // ---------------- depthwise conv -> lA ----------------
  {
    const int c0 = tid * 2;
    float2 wr[6], wo[6], wa[6];
#pragma unroll
    for (int k = 0; k < 6; ++k) {
      wr[k] = make_float2(w_rtg[c0 * 6 + k], w_rtg[(c0 + 1) * 6 + k]);
      wo[k] = make_float2(w_obs[c0 * 6 + k], w_obs[(c0 + 1) * 6 + k]);
      wa[k] = make_float2(w_act[c0 * 6 + k], w_act[(c0 + 1) * 6 + k]);
    }
    const float2 br = make_float2(b_rtg[c0], b_rtg[c0 + 1]);
    const float2 bo = make_float2(b_obs[c0], b_obs[c0 + 1]);
    const float2 ba = make_float2(b_act[c0], b_act[c0 + 1]);

    const float* xp = x + (size_t)t0 * kC + c0;
    float2 win[6];                             // sliding window x[t-5..t]
    if (tloc == 0) {
#pragma unroll
      for (int k = 1; k < 6; ++k) win[k] = make_float2(0.f, 0.f);
    } else {
#pragma unroll
      for (int k = 1; k < 6; ++k) {
        const float* p = xp + (ptrdiff_t)(k - 6) * kC;
        win[k] = make_float2(p[0], p[1]);
      }
    }

    int g = tloc % 3;
    for (int r = 0; r < M_BLK; ++r) {
#pragma unroll
      for (int k = 0; k < 5; ++k) win[k] = win[k + 1];
      const float* p = xp + (size_t)r * kC;
      win[5] = make_float2(p[0], p[1]);        // each x element loaded exactly once

      float2 s;
      if (g == 0) {
        s = br;
#pragma unroll
        for (int k = 0; k < 6; ++k) { s.x = fmaf(win[k].x, wr[k].x, s.x); s.y = fmaf(win[k].y, wr[k].y, s.y); }
      } else if (g == 1) {
        s = bo;
#pragma unroll
        for (int k = 0; k < 6; ++k) { s.x = fmaf(win[k].x, wo[k].x, s.x); s.y = fmaf(win[k].y, wo[k].y, s.y); }
      } else {
        s = ba;
#pragma unroll
        for (int k = 0; k < 6; ++k) { s.x = fmaf(win[k].x, wa[k].x, s.x); s.y = fmaf(win[k].y, wa[k].y, s.y); }
      }
      g = (g == 2) ? 0 : g + 1;

      union { _Float16 h[2]; unsigned int u32; } pk;
      pk.h[0] = (_Float16)s.x; pk.h[1] = (_Float16)s.y;
      *reinterpret_cast<unsigned int*>(&lA[r * LDA + c0]) = pk.u32;
    }
  }

  // ---------------- WMMA GEMM with async double-buffered B ----------------
  const int lane  = tid & 31;
  const int ln15  = lane & 15;
  const int khalf = lane >> 4;
  const int wv    = tid >> 5;
  const int wvm   = wv >> 1;                   // 0..3 : 32-row group
  const int wvn   = wv & 1;                    // 0..1 : 64-col group

  // A frag (ISA 16-bit A 16x32): two 8-half chunks at K = khalf*8 and 16+khalf*8
  const _Float16* pA = lA + (wvm * 32 + ln15) * LDA + khalf * 8;
  // B frag (ISA 16-bit B 32x16): two 8-half chunks at K = khalf*8 and 16+khalf*8
  const int pBoff = (wvn * 64 + ln15) * LDB + khalf * 8;

  v8f acc[2][4];

  for (int s = 0; s < N_SEG; ++s) {            // segment = (nc = s>>1, kb = s&1)
    const int nc = s >> 1;
    const int kb = s & 1;

    if (s + 1 < N_SEG) {                       // prefetch next slice into other buffer
      const int nc2 = (s + 1) >> 1;
      const int kb2 = (s + 1) & 1;
      stage_b_async(wf16 + ((size_t)nc2 * N_CHUNK) * kC + kb2 * K_HALF,
                    ((s + 1) & 1) ? lB1 : lB0, tid);
      // 16 loads in flight for next slice; in-order completion => current slice landed
      asm volatile("s_wait_asynccnt 16" ::: "memory");
    } else {
      asm volatile("s_wait_asynccnt 0" ::: "memory");
    }
    __syncthreads();                           // slice visible to all waves (also guards lA on s==0)

    const _Float16* pB = ((s & 1) ? lB1 : lB0) + pBoff;

    if (kb == 0) {
#pragma unroll
      for (int mi = 0; mi < 2; ++mi)
#pragma unroll
        for (int nt = 0; nt < 4; ++nt)
#pragma unroll
          for (int e = 0; e < 8; ++e) acc[mi][nt][e] = 0.f;
    }

#pragma unroll 4
    for (int kcl = 0; kcl < 8; ++kcl) {        // global K step = kb*8 + kcl
      const int kA = (kb * 8 + kcl) * 32;
      const v8h a0l = *reinterpret_cast<const v8h*>(pA + kA);
      const v8h a0h = *reinterpret_cast<const v8h*>(pA + kA + 16);
      const v8h a1l = *reinterpret_cast<const v8h*>(pA + 16 * LDA + kA);
      const v8h a1h = *reinterpret_cast<const v8h*>(pA + 16 * LDA + kA + 16);
      const v16h a0 = __builtin_shufflevector(a0l, a0h, 0,1,2,3,4,5,6,7,8,9,10,11,12,13,14,15);
      const v16h a1 = __builtin_shufflevector(a1l, a1h, 0,1,2,3,4,5,6,7,8,9,10,11,12,13,14,15);
#pragma unroll
      for (int nt = 0; nt < 4; ++nt) {
        const v8h b_lo = *reinterpret_cast<const v8h*>(pB + nt * 16 * LDB + kcl * 32);
        const v8h b_hi = *reinterpret_cast<const v8h*>(pB + nt * 16 * LDB + kcl * 32 + 16);
        const v16h b = __builtin_shufflevector(b_lo, b_hi, 0,1,2,3,4,5,6,7,8,9,10,11,12,13,14,15);
        acc[0][nt] = __builtin_amdgcn_wmma_f32_16x16x32_f16(false, a0, false, b,
                                                            (short)0, acc[0][nt], false, false);
        acc[1][nt] = __builtin_amdgcn_wmma_f32_16x16x32_f16(false, a1, false, b,
                                                            (short)0, acc[1][nt], false, false);
      }
    }

    if (kb == 1) {                             // K complete for this nc: bias + store fp32
#pragma unroll
      for (int mi = 0; mi < 2; ++mi) {
#pragma unroll
        for (int nt = 0; nt < 4; ++nt) {
          const int col    = nc * N_CHUNK + wvn * 64 + nt * 16 + ln15;
          const float bias = fc_b[col];
          float* po = out + ((size_t)t0 + wvm * 32 + mi * 16 + khalf * 8) * kC + col;
#pragma unroll
          for (int e = 0; e < 8; ++e) po[(size_t)e * kC] = acc[mi][nt][e] + bias;
        }
      }
    }
    __syncthreads();                           // all waves done with this buffer
  }
}

extern "C" void kernel_launch(void* const* d_in, const int* in_sizes, int n_in,
                              void* d_out, int out_size, void* d_ws, size_t ws_size,
                              hipStream_t stream) {
  const float* x     = (const float*)d_in[0];
  const float* w_rtg = (const float*)d_in[1];
  const float* b_rtg = (const float*)d_in[2];
  const float* w_obs = (const float*)d_in[3];
  const float* b_obs = (const float*)d_in[4];
  const float* w_act = (const float*)d_in[5];
  const float* b_act = (const float*)d_in[6];
  const float* fc_w  = (const float*)d_in[7];
  const float* fc_b  = (const float*)d_in[8];
  float* out = (float*)d_out;
  _Float16* wf16 = (_Float16*)d_ws;                         // 512KB scratch

  cvt_fcw_f16<<<(kC * kC) / 256, 256, 0, stream>>>(fc_w, wf16);

  const int nblocks  = (kB * kT) / M_BLK;                   // 768
  const size_t shmem = (size_t)(M_BLK * LDA + 2 * N_CHUNK * LDB) * sizeof(_Float16); // 268288
  fused_dwconv_gemm<<<nblocks, 256, shmem, stream>>>(
      x, w_rtg, b_rtg, w_obs, b_obs, w_act, b_act, wf16, fc_b, out);
}
